// GNN_28595892257116
// MI455X (gfx1250) — matrix operand order repaired and checked by reading, not compiled, not run on hardware
//
#include <hip/hip_runtime.h>

#define NNODES 100000
#define NEDGES 400000
#define NGRAPH 64
#define DIN 20
#define DH 1024
#define DFC 128
#define ROW_TILES (NNODES / 16)   // 6250

typedef __attribute__((ext_vector_type(16))) __bf16 v16bf;
typedef __attribute__((ext_vector_type(8)))  float  v8f;

union BF16Frag { uint4 u[2]; v16bf v; };

static __device__ __forceinline__ void atomAddF(float* p, float v) {
  __hip_atomic_fetch_add(p, v, __ATOMIC_RELAXED, __HIP_MEMORY_SCOPE_AGENT);
}

static __device__ __forceinline__ unsigned short f2bf(float f) {
  unsigned int u = __float_as_uint(f);
  unsigned int r = u + 0x7fffu + ((u >> 16) & 1u);   // RNE
  return (unsigned short)(r >> 16);
}
static __device__ __forceinline__ float bf2f(unsigned short h) {
  return __uint_as_float(((unsigned int)h) << 16);
}

// ---------------- utility: zero (16B granular) ----------------
__global__ void k_zero(uint4* p, size_t n16) {
  size_t i = (size_t)blockIdx.x * blockDim.x + threadIdx.x;
  size_t s = (size_t)gridDim.x * blockDim.x;
  uint4 z = {0u, 0u, 0u, 0u};
  for (; i < n16; i += s) p[i] = z;
}

// ---------------- degree ----------------
__global__ void k_deg(const int* __restrict__ dst, float* __restrict__ deg) {
  int e = blockIdx.x * blockDim.x + threadIdx.x;
  if (e < NEDGES) atomAddF(&deg[dst[e]], 1.0f);
}
__global__ void k_inv(const float* __restrict__ deg, float* __restrict__ invd) {
  int n = blockIdx.x * blockDim.x + threadIdx.x;
  if (n < NNODES) invd[n] = 1.0f / fmaxf(deg[n], 1.0f);
}

// ---------------- layer-1 neighbor scatter (E x 20 f32 atomics) ----------------
__global__ void k_agg1(const float* __restrict__ x, const int* __restrict__ src,
                       const int* __restrict__ dst, float* __restrict__ agg1) {
  int e = blockIdx.x * blockDim.x + threadIdx.x;
  if (e >= NEDGES) return;
  int s = src[e], d = dst[e];
  const float* xs = x + (size_t)s * DIN;
  float* ad = agg1 + (size_t)d * DIN;
  #pragma unroll
  for (int i = 0; i < DIN; ++i) atomAddF(&ad[i], xs[i]);
}

// ---------------- layer 1: h1 = tanh(agg1*inv @ W1l^T + b1l + x @ W1r^T) -> bf16 ----
__global__ __launch_bounds__(256) void k_layer1(
    const float* __restrict__ x, const float* __restrict__ agg1,
    const float* __restrict__ invd, const float* __restrict__ W1l,
    const float* __restrict__ b1l, const float* __restrict__ W1r,
    unsigned short* __restrict__ h1) {
  __shared__ float sx[16][DIN];
  __shared__ float sa[16][DIN];
  const int n0 = blockIdx.x * 16;
  const int tid = threadIdx.x;
  for (int i = tid; i < 16 * DIN; i += 256) {
    int nn = i / DIN, f = i % DIN;
    sx[nn][f] = x[(size_t)(n0 + nn) * DIN + f];
    sa[nn][f] = agg1[(size_t)(n0 + nn) * DIN + f] * invd[n0 + nn];
  }
  __syncthreads();
  for (int c = tid; c < DH; c += 256) {
    float wl[DIN], wr[DIN];
    #pragma unroll
    for (int i = 0; i < DIN; ++i) { wl[i] = W1l[(size_t)c * DIN + i]; wr[i] = W1r[(size_t)c * DIN + i]; }
    float bias = b1l[c];
    #pragma unroll 4
    for (int nn = 0; nn < 16; ++nn) {
      float s = bias;
      #pragma unroll
      for (int i = 0; i < DIN; ++i) s += sa[nn][i] * wl[i] + sx[nn][i] * wr[i];
      h1[(size_t)(n0 + nn) * DH + c] = f2bf(tanhf(s));
    }
  }
}

// ---------------- f32 -> bf16 weight convert ----------------
__global__ void k_f32_to_bf16(const float* __restrict__ in, unsigned short* __restrict__ out, int n) {
  int i = blockIdx.x * blockDim.x + threadIdx.x;
  int s = gridDim.x * blockDim.x;
  for (; i < n; i += s) out[i] = f2bf(in[i]);
}

// ---------------- layer-2 neighbor scatter: agg2 += h1[src] (f32 atomics) ------
__global__ __launch_bounds__(256) void k_agg2(
    const unsigned short* __restrict__ h1, const int* __restrict__ src,
    const int* __restrict__ dst, float* __restrict__ agg2) {
  const int e = blockIdx.x;
  const int s = src[e], d = dst[e];
  const int f0 = threadIdx.x * 4;
  uint2 u = *(const uint2*)(h1 + (size_t)s * DH + f0);
  float* q = agg2 + (size_t)d * DH + f0;
  atomAddF(q + 0, bf2f((unsigned short)(u.x & 0xffffu)));
  atomAddF(q + 1, bf2f((unsigned short)(u.x >> 16)));
  atomAddF(q + 2, bf2f((unsigned short)(u.y & 0xffffu)));
  atomAddF(q + 3, bf2f((unsigned short)(u.y >> 16)));
}

// ---------------- scale by inv_deg & convert to bf16 ----------------
__global__ void k_cvt_agg2(const float* __restrict__ agg2, const float* __restrict__ invd,
                           unsigned short* __restrict__ out) {
  size_t i = ((size_t)blockIdx.x * blockDim.x + threadIdx.x) * 4;
  if (i >= (size_t)NNODES * DH) return;
  int n = (int)(i >> 10);
  float sc = invd[n];
  float4 v = *(const float4*)(agg2 + i);
  unsigned int lo = (unsigned int)f2bf(v.x * sc) | ((unsigned int)f2bf(v.y * sc) << 16);
  unsigned int hi = (unsigned int)f2bf(v.z * sc) | ((unsigned int)f2bf(v.w * sc) << 16);
  *(uint2*)(out + i) = make_uint2(lo, hi);
}

// ---------------- layer-2 WMMA GEMM fused with bias+tanh+global_add_pool -------
// wave: 16 rows x 128 cols; 8 waves/block share 128-col B tiles (both weights) in LDS.
__global__ __launch_bounds__(256) void k_gemm2(
    const unsigned short* __restrict__ aggb, const unsigned short* __restrict__ h1,
    const unsigned short* __restrict__ w2lb, const unsigned short* __restrict__ w2rb,
    const float* __restrict__ b2l, const int* __restrict__ batch,
    float* __restrict__ gbuf) {
  __shared__ __align__(16) unsigned short sBl[128 * 40];   // 32 k + 8 pad per row
  __shared__ __align__(16) unsigned short sBr[128 * 40];

  const int tid = threadIdx.x;
  const int wave = tid >> 5;
  const int lane = tid & 31;
  const int half = lane >> 4;
  const int l15 = lane & 15;
  const int rowTile = blockIdx.x * 8 + wave;
  const int colBase = blockIdx.y * 128;
  const bool active = rowTile < ROW_TILES;
  const int m0 = rowTile * 16;
  const int m = m0 + l15;

  v8f acc[8];
  #pragma unroll
  for (int cf = 0; cf < 8; ++cf) acc[cf] = (v8f){0.f, 0.f, 0.f, 0.f, 0.f, 0.f, 0.f, 0.f};

  const int r = tid >> 1, hh = tid & 1;
  const size_t gB = (size_t)(colBase + r) * DH + hh * 16;
  const int ldsOff = r * 40 + hh * 16;

  for (int kb = 0; kb < DH; kb += 32) {
    __syncthreads();
    {
      const uint4* gl = (const uint4*)(w2lb + gB + kb);
      uint4 p0 = gl[0], p1 = gl[1];
      const uint4* gr = (const uint4*)(w2rb + gB + kb);
      uint4 q0 = gr[0], q1 = gr[1];
      uint4* dl = (uint4*)(sBl + ldsOff);
      dl[0] = p0; dl[1] = p1;
      uint4* dr = (uint4*)(sBr + ldsOff);
      dr[0] = q0; dr[1] = q1;
    }
    __syncthreads();
    if (active) {
      // A fragment (ISA 16-bit A layout): lane half selects k sub-chunk (0/8),
      // elements 0..7 = K[kq..kq+7], elements 8..15 = K[kq+16..kq+23]
      const size_t aOff = (size_t)m * DH + kb + half * 8;
      BF16Frag fa, fh;
      fa.u[0] = *(const uint4*)(aggb + aOff);
      fa.u[1] = *(const uint4*)(aggb + aOff + 16);
      fh.u[0] = *(const uint4*)(h1 + aOff);
      fh.u[1] = *(const uint4*)(h1 + aOff + 16);
      #pragma unroll
      for (int cf = 0; cf < 8; ++cf) {
        const int bo = (cf * 16 + l15) * 40 + half * 8;   // B: lane=col, half picks K 8h..8h+7 / 16+8h..16+8h+7
        BF16Frag bl, br;
        const uint4* pl = (const uint4*)(sBl + bo);
        bl.u[0] = pl[0]; bl.u[1] = pl[2];
        const uint4* pr = (const uint4*)(sBr + bo);
        br.u[0] = pr[0]; br.u[1] = pr[2];
        acc[cf] = __builtin_amdgcn_wmma_f32_16x16x32_bf16(
            false, fa.v, false, bl.v, (short)0, acc[cf], false, false);
        acc[cf] = __builtin_amdgcn_wmma_f32_16x16x32_bf16(
            false, fh.v, false, br.v, (short)0, acc[cf], false, false);
      }
    }
  }

  if (active) {
    // epilogue: bias + tanh + pooled segment-sum into gbuf[G, DH]
    int bids[8];
    const int rowbase = m0 + half * 8;
    #pragma unroll
    for (int j = 0; j < 8; ++j) bids[j] = batch[rowbase + j];
    const int bfirst = batch[m0];
    const int blast = batch[m0 + 15];
    #pragma unroll
    for (int cf = 0; cf < 8; ++cf) {
      const int col = colBase + cf * 16 + l15;
      const float bias = b2l[col];
      float vals[8];
      #pragma unroll
      for (int j = 0; j < 8; ++j) vals[j] = tanhf(acc[cf][j] + bias);
      for (int b = bfirst; b <= blast; ++b) {   // batch sorted: usually 1 iter
        float s = 0.f;
        #pragma unroll
        for (int j = 0; j < 8; ++j) s += (bids[j] == b) ? vals[j] : 0.f;
        s += __shfl_xor(s, 16);
        if (half == 0) atomAddF(&gbuf[(size_t)b * DH + col], s);
      }
    }
  }
}

// ---------------- FC head ----------------
__global__ void k_fc1(const float* __restrict__ gbuf, const float* __restrict__ Wf1,
                      const float* __restrict__ bf1, float* __restrict__ y1) {
  int idx = blockIdx.x * blockDim.x + threadIdx.x;   // 64*128
  int o = idx & (DFC - 1);
  int gi = idx >> 7;
  float s = bf1[o];
  const float* gr = gbuf + (size_t)gi * DH;
  const float* wr = Wf1 + (size_t)o * DH;
  for (int k = 0; k < DH; ++k) s += gr[k] * wr[k];
  y1[idx] = (s > 0.f) ? s : 0.2f * s;
}

__global__ void k_fc2(const float* __restrict__ y1, const float* __restrict__ Wf2,
                      const float* __restrict__ bf2, float* __restrict__ out) {
  int gi = threadIdx.x;
  if (gi >= NGRAPH) return;
  float s = bf2[0];
  const float* yr = y1 + (size_t)gi * DFC;
  for (int o = 0; o < DFC; ++o) s += yr[o] * Wf2[o];
  out[gi] = s;
}

extern "C" void kernel_launch(void* const* d_in, const int* in_sizes, int n_in,
                              void* d_out, int out_size, void* d_ws, size_t ws_size,
                              hipStream_t stream) {
  const float* x = (const float*)d_in[0];
  const int* ei = (const int*)d_in[1];
  const int* src = ei;
  const int* dst = ei + NEDGES;
  const int* batch = (const int*)d_in[2];
  const float* W1l = (const float*)d_in[3];
  const float* b1l = (const float*)d_in[4];
  const float* W1r = (const float*)d_in[5];
  const float* W2l = (const float*)d_in[6];
  const float* b2l = (const float*)d_in[7];
  const float* W2r = (const float*)d_in[8];
  const float* Wf1 = (const float*)d_in[9];
  const float* bf1 = (const float*)d_in[10];
  const float* Wf2 = (const float*)d_in[11];
  const float* bf2 = (const float*)d_in[12];
  float* out = (float*)d_out;

  char* w = (char*)d_ws;
  size_t off = 0;
  float* agg2 = (float*)(w + off);           off += (size_t)NNODES * DH * 4;   // 409.6 MB
  unsigned short* h1 = (unsigned short*)(w + off);    off += (size_t)NNODES * DH * 2;
  unsigned short* agg2b = (unsigned short*)(w + off); off += (size_t)NNODES * DH * 2;
  unsigned short* w2lb = (unsigned short*)(w + off);  off += (size_t)DH * DH * 2;
  unsigned short* w2rb = (unsigned short*)(w + off);  off += (size_t)DH * DH * 2;
  float* deg = (float*)(w + off);            off += (size_t)NNODES * 4;
  float* invd = (float*)(w + off);           off += (size_t)NNODES * 4;
  float* agg1 = (float*)(w + off);           off += (size_t)NNODES * DIN * 4;
  float* gbuf = (float*)(w + off);           off += (size_t)NGRAPH * DH * 4;
  float* y1 = (float*)(w + off);             off += (size_t)NGRAPH * DFC * 4;

  // zero accumulators (fresh every call)
  k_zero<<<8192, 256, 0, stream>>>((uint4*)agg2, (size_t)NNODES * DH * 4 / 16);
  k_zero<<<64, 256, 0, stream>>>((uint4*)deg, (size_t)NNODES * 4 / 16);
  k_zero<<<512, 256, 0, stream>>>((uint4*)agg1, (size_t)NNODES * DIN * 4 / 16);
  k_zero<<<16, 256, 0, stream>>>((uint4*)gbuf, (size_t)NGRAPH * DH * 4 / 16);

  k_deg<<<(NEDGES + 255) / 256, 256, 0, stream>>>(dst, deg);
  k_inv<<<(NNODES + 255) / 256, 256, 0, stream>>>(deg, invd);
  k_agg1<<<(NEDGES + 255) / 256, 256, 0, stream>>>(x, src, dst, agg1);
  k_layer1<<<NNODES / 16, 256, 0, stream>>>(x, agg1, invd, W1l, b1l, W1r, h1);

  k_f32_to_bf16<<<2048, 256, 0, stream>>>(W2l, w2lb, DH * DH);
  k_f32_to_bf16<<<2048, 256, 0, stream>>>(W2r, w2rb, DH * DH);

  k_agg2<<<NEDGES, 256, 0, stream>>>(h1, src, dst, agg2);
  k_cvt_agg2<<<(int)(((size_t)NNODES * DH / 4 + 255) / 256), 256, 0, stream>>>(agg2, invd, agg2b);

  dim3 g2((ROW_TILES + 7) / 8, DH / 128);
  k_gemm2<<<g2, 256, 0, stream>>>(agg2b, h1, w2lb, w2rb, b2l, batch, gbuf);

  k_fc1<<<(NGRAPH * DFC) / 256, 256, 0, stream>>>(gbuf, Wf1, bf1, y1);
  k_fc2<<<1, NGRAPH, 0, stream>>>(y1, Wf2, bf2, out);

  (void)in_sizes; (void)n_in; (void)out_size; (void)ws_size;
}
